// ImprovedModel_56014963474649
// MI455X (gfx1250) — matrix operand order repaired and checked by reading, not compiled, not run on hardware
//
#include <hip/hip_runtime.h>
#include <hip/hip_bf16.h>
#include <cstddef>
#include <cstdint>

typedef __attribute__((ext_vector_type(16))) _Float16 v16h;
typedef __attribute__((ext_vector_type(8)))  _Float16 v8h;
typedef __attribute__((ext_vector_type(4)))  _Float16 v4h;
typedef __attribute__((ext_vector_type(8)))  float    v8f;

#define EPSBN 1e-5f

// ---------------------------------------------------------------- utilities
__device__ __forceinline__ float sigm(float x) { return 1.0f / (1.0f + expf(-x)); }
__device__ __forceinline__ float gelu_exact(float x) {
    return 0.5f * x * (1.0f + erff(x * 0.70710678118654752f));
}
// monotonic float<->uint mapping for atomicMax-based segment max
__device__ __forceinline__ unsigned fenc(float f) {
    unsigned u = __float_as_uint(f);
    return (u & 0x80000000u) ? ~u : (u | 0x80000000u);
}
__device__ __forceinline__ float fdec(unsigned u) {
    unsigned b = (u & 0x80000000u) ? (u & 0x7FFFFFFFu) : ~u;
    return __uint_as_float(b);
}

// ---------------------------------------------------------------- conversions
__global__ void cvt_f32_f16(const float* __restrict__ in, _Float16* __restrict__ out, int n) {
    int i = blockIdx.x * blockDim.x + threadIdx.x;
    if (i < n) out[i] = (_Float16)in[i];
}

__global__ void add_vec(const float* __restrict__ a, const float* __restrict__ b,
                        float* __restrict__ o, int n) {
    int i = blockIdx.x * blockDim.x + threadIdx.x;
    if (i < n) o[i] = a[i] + b[i];
}

// ---------------------------------------------------------------- WMMA GEMM
// C[M,N] = A[M,K] @ W[N,K]^T (+bias[N]); A,W f16 row-major, C f32.
// Block: 256 threads = 8 waves (2 waveM x 4 waveN); block tile 32x128,
// each wave 16x32 (2 accumulators). Double-buffered LDS staging.
// M % 32 == 0, N % 128 == 0, K % 32 == 0 (true for all call sites).
#define TILE_M 32
#define TILE_N 128
#define KSTEP  32
#define LDPAD  8   // pad to 40 halfs = 80B rows (multiple of 16B, bank-skewed)

__global__ __launch_bounds__(256)
void gemm_f16_wmma(const _Float16* __restrict__ A, const _Float16* __restrict__ W,
                   float* __restrict__ C, const float* __restrict__ bias,
                   int M, int N, int K) {
    __shared__ _Float16 As[2][TILE_M][KSTEP + LDPAD];
    __shared__ _Float16 Bs[2][TILE_N][KSTEP + LDPAD];   // W rows staged untransposed

    const int tid   = threadIdx.x;
    const int wave  = tid >> 5;      // 0..7
    const int lane  = tid & 31;
    const int waveM = wave >> 2;     // 0..1 -> 16-row slabs
    const int waveN = wave & 3;      // 0..3 -> 32-col slabs
    const int lm    = lane & 15;
    const int hi    = lane >> 4;

    const int rowBlock = blockIdx.x * TILE_M;
    const int colBlock = blockIdx.y * TILE_N;

    // staging coordinates
    const int ar = tid >> 3;              // 0..31  A row
    const int ac = (tid & 7) * 4;         // 0..28  A col (4 halfs)
    const int bn = tid >> 1;              // 0..127 B row (n)
    const int bk = (tid & 1) * 16;        // 0 / 16 B col (16 halfs)

    auto stage = [&](int kk, int buf) {
        *(v4h*)&As[buf][ar][ac] = *(const v4h*)(A + (size_t)(rowBlock + ar) * K + kk + ac);
        const _Float16* s = W + (size_t)(colBlock + bn) * K + kk + bk;
        *(v8h*)&Bs[buf][bn][bk]     = *(const v8h*)(s);
        *(v8h*)&Bs[buf][bn][bk + 8] = *(const v8h*)(s + 8);
    };

    v8f acc0 = {}, acc1 = {};

    stage(0, 0);
    __syncthreads();

    const int nk = K / KSTEP;
    for (int i = 0; i < nk; ++i) {
        const int cur = i & 1;
        if (i + 1 < nk) stage((i + 1) * KSTEP, cur ^ 1);

        // A fragment: per-lane k-runs are contiguous -> 2x ds_load_b128
        const _Float16* arow = &As[cur][waveM * 16 + lm][0];
        v8h a0 = *(const v8h*)(arow + hi * 16);
        v8h a1 = *(const v8h*)(arow + hi * 16 + 8);
        v16h af = __builtin_shufflevector(a0, a1, 0, 1, 2, 3, 4, 5, 6, 7,
                                          8, 9, 10, 11, 12, 13, 14, 15);
        // B fragments: element e = W[n][kk + hi*16 + e] -> contiguous 32B per lane
        const _Float16* brow0 = &Bs[cur][waveN * 32 + lm][hi * 16];
        v8h b00 = *(const v8h*)(brow0);
        v8h b01 = *(const v8h*)(brow0 + 8);
        v16h bf0 = __builtin_shufflevector(b00, b01, 0, 1, 2, 3, 4, 5, 6, 7,
                                           8, 9, 10, 11, 12, 13, 14, 15);
        const _Float16* brow1 = &Bs[cur][waveN * 32 + 16 + lm][hi * 16];
        v8h b10 = *(const v8h*)(brow1);
        v8h b11 = *(const v8h*)(brow1 + 8);
        v16h bf1 = __builtin_shufflevector(b10, b11, 0, 1, 2, 3, 4, 5, 6, 7,
                                           8, 9, 10, 11, 12, 13, 14, 15);

        acc0 = __builtin_amdgcn_wmma_f32_16x16x32_f16(false, af, false, bf0,
                                                      (short)0, acc0, false, false);
        acc1 = __builtin_amdgcn_wmma_f32_16x16x32_f16(false, af, false, bf1,
                                                      (short)0, acc1, false, false);
        __syncthreads();
    }

    const int col0 = colBlock + waveN * 32 + lm;
    const int col1 = col0 + 16;
    const float b0 = bias ? bias[col0] : 0.0f;
    const float b1 = bias ? bias[col1] : 0.0f;
#pragma unroll
    for (int r = 0; r < 8; ++r) {
        int row = rowBlock + waveM * 16 + r + hi * 8;   // C/D layout: VGPR r -> M=r(+8)
        C[(size_t)row * N + col0] = acc0[r] + b0;
        C[(size_t)row * N + col1] = acc1[r] + b1;
    }
}

// ---------------------------------------------------------------- GAT edge ops
__global__ void build_edges(const int* __restrict__ ei, int* __restrict__ srcv,
                            int* __restrict__ dstv, int E, int nloops) {
    int i = blockIdx.x * blockDim.x + threadIdx.x;
    int tot = E + nloops;
    if (i >= tot) return;
    if (i < E) { srcv[i] = ei[i]; dstv[i] = ei[E + i]; }
    else       { srcv[i] = i - E; dstv[i] = i - E; }
}

__global__ void gat_logits(const float* __restrict__ xl, const float* __restrict__ xr,
                           const float* __restrict__ att, const int* __restrict__ srcv,
                           const int* __restrict__ dstv, float* __restrict__ logit,
                           unsigned* __restrict__ mmax, int B, int Etot, int NH, int NN) {
    int idx = blockIdx.x * blockDim.x + threadIdx.x;
    int total = B * Etot * NH;
    if (idx >= total) return;
    int h = idx % NH;
    int e = (idx / NH) % Etot;
    int g = idx / (NH * Etot);
    int s = srcv[e], d0 = dstv[e];
    const float* pl = xl + ((size_t)(g * NN + s)  * NH + h) * 128;
    const float* pr = xr + ((size_t)(g * NN + d0) * NH + h) * 128;
    const float* pa = att + h * 128;
    float acc = 0.0f;
    for (int d = 0; d < 128; ++d) {
        float v = pl[d] + pr[d];
        v = v > 0.0f ? v : 0.2f * v;        // leaky_relu(0.2)
        acc += v * pa[d];
    }
    logit[idx] = acc;
    atomicMax(&mmax[((size_t)g * NN + d0) * NH + h], fenc(acc));
}

__global__ void gat_expsum(float* __restrict__ logit, const unsigned* __restrict__ mmax,
                           float* __restrict__ ssum, const int* __restrict__ dstv,
                           int B, int Etot, int NH, int NN) {
    int idx = blockIdx.x * blockDim.x + threadIdx.x;
    int total = B * Etot * NH;
    if (idx >= total) return;
    int h = idx % NH;
    int e = (idx / NH) % Etot;
    int g = idx / (NH * Etot);
    int d0 = dstv[e];
    float p = expf(logit[idx] - fdec(mmax[((size_t)g * NN + d0) * NH + h]));
    logit[idx] = p;
    atomicAdd(&ssum[((size_t)g * NN + d0) * NH + h], p);
}

__global__ void gat_aggregate(const float* __restrict__ xl, const float* __restrict__ p,
                              const float* __restrict__ ssum, const int* __restrict__ srcv,
                              const int* __restrict__ dstv, float* __restrict__ out,
                              int B, int Etot, int NH, int NN) {
    int idx = blockIdx.x * blockDim.x + threadIdx.x;
    if (idx >= B * Etot) return;
    int e = idx % Etot;
    int g = idx / Etot;
    int s = srcv[e], d0 = dstv[e];
    for (int h = 0; h < NH; ++h) {
        float alpha = p[((size_t)g * Etot + e) * NH + h] /
                      ssum[((size_t)g * NN + d0) * NH + h];
        const float* pl = xl + ((size_t)(g * NN + s)  * NH + h) * 128;
        float*       po = out + ((size_t)(g * NN + d0) * NH + h) * 128;
        for (int d = 0; d < 128; ++d) atomicAdd(&po[d], alpha * pl[d]);
    }
}

// elu(bn(in + bg)) -> f16
__global__ void bn_elu_to_f16(const float* __restrict__ in, const float* __restrict__ bg,
                              const float* __restrict__ ga, const float* __restrict__ be,
                              const float* __restrict__ mn, const float* __restrict__ vr,
                              _Float16* __restrict__ out16, int total, int C) {
    int i = blockIdx.x * blockDim.x + threadIdx.x;
    if (i >= total) return;
    int c = i % C;
    float v = in[i] + bg[c];
    v = (v - mn[c]) * (ga[c] * rsqrtf(vr[c] + EPSBN)) + be[c];
    v = v > 0.0f ? v : expm1f(v);
    out16[i] = (_Float16)v;
}

// elu(bn(in + bg)) + proj + bp -> f16
__global__ void bn_elu_residual_f16(const float* __restrict__ in, const float* __restrict__ proj,
                                    const float* __restrict__ bg, const float* __restrict__ ga,
                                    const float* __restrict__ be, const float* __restrict__ mn,
                                    const float* __restrict__ vr, const float* __restrict__ bp,
                                    _Float16* __restrict__ out16, int total, int C) {
    int i = blockIdx.x * blockDim.x + threadIdx.x;
    if (i >= total) return;
    int c = i % C;
    float v = in[i] + bg[c];
    v = (v - mn[c]) * (ga[c] * rsqrtf(vr[c] + EPSBN)) + be[c];
    v = v > 0.0f ? v : expm1f(v);
    v += proj[i] + bp[c];
    out16[i] = (_Float16)v;
}

// ---------------------------------------------------------------- LSTM gates
__global__ void lstm_gate(const float* __restrict__ pre, const float* __restrict__ gbuf,
                          float* __restrict__ cstate, _Float16* __restrict__ hstate16,
                          float* __restrict__ out_f32, _Float16* __restrict__ out_f16, int ti) {
    int idx = blockIdx.x * blockDim.x + threadIdx.x;
    if (idx >= 4000 * 128) return;
    int r = idx >> 7;
    int d = idx & 127;
    size_t prow = ((size_t)(r * 16 + ti)) * 512;
    size_t grow = (size_t)r * 512;
    float i_ = pre[prow + d]       + gbuf[grow + d];
    float f_ = pre[prow + 128 + d] + gbuf[grow + 128 + d];
    float g_ = pre[prow + 256 + d] + gbuf[grow + 256 + d];
    float o_ = pre[prow + 384 + d] + gbuf[grow + 384 + d];
    float c  = sigm(f_) * cstate[idx] + sigm(i_) * tanhf(g_);
    float h  = sigm(o_) * tanhf(c);
    cstate[idx]   = c;
    hstate16[idx] = (_Float16)h;
    size_t orow = ((size_t)(r * 16 + ti)) * 128 + d;
    if (out_f32) out_f32[orow] = h;
    if (out_f16) out_f16[orow] = (_Float16)h;
}

// ---------------------------------------------------------------- attention pool
__global__ __launch_bounds__(128)
void attn_pool(const float* __restrict__ lo, const float* __restrict__ Wa,
               const float* __restrict__ ba, float* __restrict__ ctx) {
    __shared__ float sc[16];
    __shared__ float wsum[4];
    const int r = blockIdx.x;
    const int d = threadIdx.x;
    const int w = d >> 5, ln = d & 31;
    const float* base = lo + (size_t)r * 16 * 128;
    for (int ti = 0; ti < 16; ++ti) {
        float v = base[ti * 128 + d] * Wa[d];
        for (int off = 16; off > 0; off >>= 1) v += __shfl_xor(v, off, 32);
        if (ln == 0) wsum[w] = v;
        __syncthreads();
        if (d == 0) sc[ti] = wsum[0] + wsum[1] + wsum[2] + wsum[3] + ba[0];
        __syncthreads();
    }
    float mx = -1e30f;
    for (int ti = 0; ti < 16; ++ti) mx = fmaxf(mx, sc[ti]);
    float aw[16], ss = 0.0f;
    for (int ti = 0; ti < 16; ++ti) { aw[ti] = expf(sc[ti] - mx); ss += aw[ti]; }
    float c = 0.0f;
    for (int ti = 0; ti < 16; ++ti) c += (aw[ti] / ss) * base[ti * 128 + d];
    ctx[(size_t)r * 128 + d] = c;
}

// ---------------------------------------------------------------- skip + head MLPs
__global__ __launch_bounds__(64)
void skip_mlp(const float* __restrict__ x, const float* __restrict__ Ws1,
              const float* __restrict__ bs1, const float* __restrict__ Ws2,
              const float* __restrict__ bs2, float* __restrict__ sk2) {
    __shared__ float hb[64];
    const int r = blockIdx.x, j = threadIdx.x;
    const int bi = r / 1000, ni = r % 1000;
    const float* xr_ = x + (((size_t)(bi * 16 + 15)) * 1000 + ni) * 32;  // x[bi,-1,ni,:]
    float acc = bs1[j];
    for (int k = 0; k < 13; ++k) acc += xr_[k] * Ws1[j * 13 + k];
    hb[j] = gelu_exact(acc);
    __syncthreads();
    if (j < 32) {
        float a2 = bs2[j];
        for (int k = 0; k < 64; ++k) a2 += hb[k] * Ws2[j * 64 + k];
        sk2[(size_t)r * 32 + j] = a2;
    }
}

__global__ __launch_bounds__(64)
void head_mlp(const float* __restrict__ ctx, const float* __restrict__ sk2,
              const float* __restrict__ Wh1, const float* __restrict__ bh1,
              const float* __restrict__ Wh2, const float* __restrict__ bh2,
              float* __restrict__ out) {
    __shared__ float red[2];
    const int r = blockIdx.x, j = threadIdx.x;
    const float* c = ctx + (size_t)r * 128;
    const float* s = sk2 + (size_t)r * 32;
    const float* wr = Wh1 + j * 160;
    float acc = bh1[j];
    for (int k = 0; k < 128; ++k) acc += c[k] * wr[k];
    for (int k = 0; k < 32;  ++k) acc += s[k] * wr[128 + k];
    acc = gelu_exact(acc) * Wh2[j];
    for (int off = 16; off > 0; off >>= 1) acc += __shfl_xor(acc, off, 32);
    if ((j & 31) == 0) red[j >> 5] = acc;
    __syncthreads();
    if (j == 0) out[r] = red[0] + red[1] + bh2[0];
}

// ---------------------------------------------------------------- host
static inline int cdiv(int a, int b) { return (a + b - 1) / b; }

extern "C" void kernel_launch(void* const* d_in, const int* in_sizes, int n_in,
                              void* d_out, int out_size, void* d_ws, size_t ws_size,
                              hipStream_t stream) {
    (void)n_in; (void)out_size; (void)ws_size;
    const float* x    = (const float*)d_in[0];
    const int*   ei   = (const int*)  d_in[1];
    const float* Wp   = (const float*)d_in[2];
    const float* bp   = (const float*)d_in[3];
    const float* Wl1  = (const float*)d_in[4];
    const float* Wr1  = (const float*)d_in[5];
    const float* a1   = (const float*)d_in[6];
    const float* bg1  = (const float*)d_in[7];
    const float* g1   = (const float*)d_in[8];
    const float* be1  = (const float*)d_in[9];
    const float* m1   = (const float*)d_in[10];
    const float* v1   = (const float*)d_in[11];
    const float* Wl2  = (const float*)d_in[12];
    const float* Wr2  = (const float*)d_in[13];
    const float* a2   = (const float*)d_in[14];
    const float* bg2  = (const float*)d_in[15];
    const float* g2   = (const float*)d_in[16];
    const float* be2  = (const float*)d_in[17];
    const float* m2   = (const float*)d_in[18];
    const float* v2   = (const float*)d_in[19];
    const float* Wih0 = (const float*)d_in[20];
    const float* Whh0 = (const float*)d_in[21];
    const float* bih0 = (const float*)d_in[22];
    const float* bhh0 = (const float*)d_in[23];
    const float* Wih1 = (const float*)d_in[24];
    const float* Whh1 = (const float*)d_in[25];
    const float* bih1 = (const float*)d_in[26];
    const float* bhh1 = (const float*)d_in[27];
    const float* Wa   = (const float*)d_in[28];
    const float* ba   = (const float*)d_in[29];
    const float* Ws1  = (const float*)d_in[30];
    const float* bs1  = (const float*)d_in[31];
    const float* Ws2  = (const float*)d_in[32];
    const float* bs2  = (const float*)d_in[33];
    const float* Wh1  = (const float*)d_in[34];
    const float* bh1  = (const float*)d_in[35];
    const float* Wh2  = (const float*)d_in[36];
    const float* bh2  = (const float*)d_in[37];
    float* out = (float*)d_out;

    const int NN = 1000, Bg = 64, M1 = 64000;
    const int E = in_sizes[1] / 2;                     // 16000
    const int Etot = E + NN;                           // 17000

    uint8_t* base = (uint8_t*)d_ws;
    size_t cur = 0;
    auto alloc = [&](size_t bytes) { void* p = base + cur; cur = (cur + bytes + 255) & ~(size_t)255; return p; };

    int*       srcv   = (int*)alloc(Etot * 4);
    int*       dstv   = (int*)alloc(Etot * 4);
    _Float16*  x16    = (_Float16*)alloc((size_t)M1 * 32 * 2);
    _Float16*  wl1_16 = (_Float16*)alloc(256 * 32 * 2);
    _Float16*  wr1_16 = (_Float16*)alloc(256 * 32 * 2);
    _Float16*  wp_16  = (_Float16*)alloc(128 * 32 * 2);
    _Float16*  wl2_16 = (_Float16*)alloc(128 * 256 * 2);
    _Float16*  wr2_16 = (_Float16*)alloc(128 * 256 * 2);
    _Float16*  wih0_16= (_Float16*)alloc(512 * 128 * 2);
    _Float16*  whh0_16= (_Float16*)alloc(512 * 128 * 2);
    _Float16*  wih1_16= (_Float16*)alloc(512 * 128 * 2);
    _Float16*  whh1_16= (_Float16*)alloc(512 * 128 * 2);
    float*     bias0  = (float*)alloc(512 * 4);
    float*     bias1  = (float*)alloc(512 * 4);
    unsigned*  mbuf1  = (unsigned*)alloc((size_t)Bg * NN * 2 * 4);
    float*     sbuf1  = (float*)alloc((size_t)Bg * NN * 2 * 4);
    unsigned*  mbuf2  = (unsigned*)alloc((size_t)Bg * NN * 4);
    float*     sbuf2  = (float*)alloc((size_t)Bg * NN * 4);
    float*     elog1  = (float*)alloc((size_t)Bg * Etot * 2 * 4);
    float*     elog2  = (float*)alloc((size_t)Bg * Etot * 4);
    float*     gbuf   = (float*)alloc((size_t)4000 * 512 * 4);
    _Float16*  hstate = (_Float16*)alloc((size_t)4000 * 128 * 2);
    float*     cstate = (float*)alloc((size_t)4000 * 128 * 4);
    float*     ctx    = (float*)alloc((size_t)4000 * 128 * 4);
    float*     sk2    = (float*)alloc((size_t)4000 * 32 * 4);
    uint8_t*   bufA   = (uint8_t*)alloc(131072000);  // xl1+xr1 | xl2+xr2+out2+proj | pre
    uint8_t*   bufB   = (uint8_t*)alloc(65536000);   // out1 | lo
    uint8_t*   bufC   = (uint8_t*)alloc(32768000);   // h1_f16 | h_f16 + seq1_f16
    float*    xl1  = (float*)(bufA);
    float*    xr1  = (float*)(bufA + 65536000);
    float*    xl2  = (float*)(bufA);
    float*    xr2  = (float*)(bufA + 32768000);
    float*    out2 = (float*)(bufA + 65536000);
    float*    proj = (float*)(bufA + 98304000);
    float*    pre  = (float*)(bufA);
    float*    out1 = (float*)(bufB);
    float*    lo   = (float*)(bufB);
    _Float16* h1_16  = (_Float16*)(bufC);
    _Float16* h_16   = (_Float16*)(bufC);
    _Float16* seq1_16= (_Float16*)(bufC + 16384000);

    auto cvt = [&](const float* s, _Float16* d, int n) {
        cvt_f32_f16<<<cdiv(n, 256), 256, 0, stream>>>(s, d, n);
    };
    auto gemm = [&](const _Float16* A, const _Float16* W, float* C, const float* bi_,
                    int M, int N, int K) {
        dim3 grid(M / TILE_M, N / TILE_N);
        gemm_f16_wmma<<<grid, 256, 0, stream>>>(A, W, C, bi_, M, N, K);
    };

    // ---- phase 0: conversions, edges, bias combine
    cvt(x, x16, M1 * 32);
    cvt(Wl1, wl1_16, 256 * 32);  cvt(Wr1, wr1_16, 256 * 32);  cvt(Wp, wp_16, 128 * 32);
    cvt(Wl2, wl2_16, 128 * 256); cvt(Wr2, wr2_16, 128 * 256);
    cvt(Wih0, wih0_16, 512 * 128); cvt(Whh0, whh0_16, 512 * 128);
    cvt(Wih1, wih1_16, 512 * 128); cvt(Whh1, whh1_16, 512 * 128);
    build_edges<<<cdiv(Etot, 256), 256, 0, stream>>>(ei, srcv, dstv, E, NN);
    add_vec<<<2, 256, 0, stream>>>(bih0, bhh0, bias0, 512);
    add_vec<<<2, 256, 0, stream>>>(bih1, bhh1, bias1, 512);

    // ---- GAT layer 1 (heads=2, d=128, concat)
    hipMemsetAsync(mbuf1, 0, (size_t)Bg * NN * 2 * 4, stream);
    hipMemsetAsync(sbuf1, 0, (size_t)Bg * NN * 2 * 4, stream);
    hipMemsetAsync(out1,  0, (size_t)M1 * 256 * 4, stream);
    gemm(x16, wl1_16, xl1, nullptr, M1, 256, 32);
    gemm(x16, wr1_16, xr1, nullptr, M1, 256, 32);
    {
        int tot = Bg * Etot * 2;
        gat_logits<<<cdiv(tot, 256), 256, 0, stream>>>(xl1, xr1, a1, srcv, dstv, elog1, mbuf1, Bg, Etot, 2, NN);
        gat_expsum<<<cdiv(tot, 256), 256, 0, stream>>>(elog1, mbuf1, sbuf1, dstv, Bg, Etot, 2, NN);
        gat_aggregate<<<cdiv(Bg * Etot, 256), 256, 0, stream>>>(xl1, elog1, sbuf1, srcv, dstv, out1, Bg, Etot, 2, NN);
    }
    bn_elu_to_f16<<<cdiv(M1 * 256, 256), 256, 0, stream>>>(out1, bg1, g1, be1, m1, v1, h1_16, M1 * 256, 256);

    // ---- GAT layer 2 (heads=1, d=128) + residual input projection
    hipMemsetAsync(mbuf2, 0, (size_t)Bg * NN * 4, stream);
    hipMemsetAsync(sbuf2, 0, (size_t)Bg * NN * 4, stream);
    hipMemsetAsync(out2,  0, (size_t)M1 * 128 * 4, stream);
    gemm(h1_16, wl2_16, xl2, nullptr, M1, 128, 256);
    gemm(h1_16, wr2_16, xr2, nullptr, M1, 128, 256);
    {
        int tot = Bg * Etot;
        gat_logits<<<cdiv(tot, 256), 256, 0, stream>>>(xl2, xr2, a2, srcv, dstv, elog2, mbuf2, Bg, Etot, 1, NN);
        gat_expsum<<<cdiv(tot, 256), 256, 0, stream>>>(elog2, mbuf2, sbuf2, dstv, Bg, Etot, 1, NN);
        gat_aggregate<<<cdiv(tot, 256), 256, 0, stream>>>(xl2, elog2, sbuf2, srcv, dstv, out2, Bg, Etot, 1, NN);
    }
    gemm(x16, wp_16, proj, nullptr, M1, 128, 32);
    bn_elu_residual_f16<<<cdiv(M1 * 128, 256), 256, 0, stream>>>(
        out2, proj, bg2, g2, be2, m2, v2, bp, h_16, M1 * 128, 128);

    // ---- LSTM layer 0 (reshape(b*n,t,H) is a flat reinterpret of [b*t,n,H])
    gemm(h_16, wih0_16, pre, bias0, M1, 512, 128);
    hipMemsetAsync(hstate, 0, (size_t)4000 * 128 * 2, stream);
    hipMemsetAsync(cstate, 0, (size_t)4000 * 128 * 4, stream);
    for (int ti = 0; ti < 16; ++ti) {
        gemm(hstate, whh0_16, gbuf, nullptr, 4000, 512, 128);
        lstm_gate<<<cdiv(4000 * 128, 256), 256, 0, stream>>>(pre, gbuf, cstate, hstate,
                                                             nullptr, seq1_16, ti);
    }
    // ---- LSTM layer 1
    gemm(seq1_16, wih1_16, pre, bias1, M1, 512, 128);
    hipMemsetAsync(hstate, 0, (size_t)4000 * 128 * 2, stream);
    hipMemsetAsync(cstate, 0, (size_t)4000 * 128 * 4, stream);
    for (int ti = 0; ti < 16; ++ti) {
        gemm(hstate, whh1_16, gbuf, nullptr, 4000, 512, 128);
        lstm_gate<<<cdiv(4000 * 128, 256), 256, 0, stream>>>(pre, gbuf, cstate, hstate,
                                                             lo, nullptr, ti);
    }

    // ---- attention pooling, skip MLP, head
    attn_pool<<<4000, 128, 0, stream>>>(lo, Wa, ba, ctx);
    skip_mlp<<<4000, 64, 0, stream>>>(x, Ws1, bs1, Ws2, bs2, sk2);
    head_mlp<<<4000, 64, 0, stream>>>(ctx, sk2, Wh1, bh1, Wh2, bh2, out);
}
